// ChunkedAttention_23639499997831
// MI455X (gfx1250) — matrix operand order repaired and checked by reading, not compiled, not run on hardware
//
#include <hip/hip_runtime.h>
#include <math.h>

// ---------------- problem constants (B,L,H,Dh,Dv)=(2,8192,16,64,64), CHUNK=1024 ----
#define B_  2
#define L_  8192
#define H_  16
#define D_  64
#define C_  1024
#define NC_ 8
#define WAVES 8            // 8 waves * 16 q-rows = 128 q-rows per workgroup
#define ROWSTRIDE (H_*D_)  // stride between consecutive sequence rows (floats)

typedef __attribute__((ext_vector_type(16))) __bf16 v16bf;
typedef __attribute__((ext_vector_type(4)))  __bf16 v4bf;
typedef __attribute__((ext_vector_type(8)))  float  v8f;
typedef __attribute__((ext_vector_type(4)))  float  v4f;

// -log(10000)/32 : inv_freq(j) = exp(j * KFREQ), j = d & 31
#define KFREQ (-0.28782313662425574f)

// ds_swizzle group-of-32: offset = (xor<<10) | (or<<5) | and(0x1F); xor16 = 0x401F
#define SWZ(v, pat) __int_as_float(__builtin_amdgcn_ds_swizzle(__float_as_int(v), (pat)))

__global__ __launch_bounds__(WAVES * 32)
void ChunkedAttention_23639499997831_kernel(const float* __restrict__ qg,
                                            const float* __restrict__ kg,
                                            const float* __restrict__ vg,
                                            const int*   __restrict__ start_index,
                                            float*       __restrict__ outg)
{
  __shared__ __bf16 lds_q [WAVES][16][64];   // per-wave RoPE'd Q tile            (16KB)
  __shared__ __bf16 lds_k [2][32][64];       // WG-shared RoPE'd K tiles [key][d]  (8KB)
  __shared__ __bf16 lds_vt[2][64][32];       // WG-shared V^T tiles [dv][key]      (8KB)

  const int tid  = threadIdx.x;
  const int lane = tid & 31;
  const int wave = tid >> 5;
  const int hh   = lane >> 4;    // 16-lane half
  const int nl   = lane & 15;

  int t = blockIdx.x;
  const int qb = t & 7;   t >>= 3;     // 8 blocks of 128 q-rows per chunk
  const int h  = t & 15;  t >>= 4;     // head
  const int n  = t & 7;   t >>= 3;     // chunk
  const int b  = t;                    // batch

  const int qrow0 = qb * 128 + wave * 16;          // first q row of this wave (in chunk)
  const int start = start_index[0];
  const int pos0  = start + n * C_;                // position of chunk row 0

  const float* qbase = qg + (((size_t)b * L_ + (size_t)n * C_ + qrow0) * H_ + h) * D_;
  const float* kbase = kg + (((size_t)b * L_ + (size_t)n * C_) * H_ + h) * D_;
  const float* vbase = vg + (((size_t)b * L_ + (size_t)n * C_) * H_ + h) * D_;

  // one inv_freq per thread (columns d=lane and d+32 share the freq index)
  const float invf = __expf((float)lane * KFREQ);

  // staging: 256 threads cover 32 rows x 32 column-pairs, 4 rows each
  const int srow0 = (tid >> 5) * 4;

  // cooperative double-buffered K/V^T staging (RoPE on K)
  auto stage_kv = [&](int buf, int key0) {
    #pragma unroll
    for (int u = 0; u < 4; ++u) {
      const int r = srow0 + u;
      const float x1 = kbase[(size_t)(key0 + r) * ROWSTRIDE + lane];
      const float x2 = kbase[(size_t)(key0 + r) * ROWSTRIDE + lane + 32];
      float s, c; __sincosf((float)(pos0 + key0 + r) * invf, &s, &c);
      lds_k[buf][r][lane]      = (__bf16)(x1 * c - x2 * s);
      lds_k[buf][r][lane + 32] = (__bf16)(x2 * c + x1 * s);
    }
    v4bf c0, c1;
    #pragma unroll
    for (int u = 0; u < 4; ++u) {
      c0[u] = (__bf16)vbase[(size_t)(key0 + srow0 + u) * ROWSTRIDE + lane];
      c1[u] = (__bf16)vbase[(size_t)(key0 + srow0 + u) * ROWSTRIDE + lane + 32];
    }
    *(v4bf*)&lds_vt[buf][lane][srow0]      = c0;
    *(v4bf*)&lds_vt[buf][lane + 32][srow0] = c1;
  };

  // ---- stage RoPE'd Q tile (16x64): lane owns columns (lane, lane+32) -----
  #pragma unroll 4
  for (int r = 0; r < 16; ++r) {
    const float x1 = qbase[(size_t)r * ROWSTRIDE + lane];
    const float x2 = qbase[(size_t)r * ROWSTRIDE + lane + 32];
    float s, c; __sincosf((float)(pos0 + qrow0 + r) * invf, &s, &c);
    lds_q[wave][r][lane]      = (__bf16)(x1 * c - x2 * s);
    lds_q[wave][r][lane + 32] = (__bf16)(x2 * c + x1 * s);
  }
  asm volatile("s_wait_dscnt 0" ::: "memory");

  // ---- Q^T as WMMA B operand (32d x 16q): lane -> q = nl, elem i -> d = hh*16+i
  v16bf bq0, bq1;
  #pragma unroll
  for (int i = 0; i < 16; ++i) {
    bq0[i] = lds_q[wave][nl][hh * 16 + i];
    bq1[i] = lds_q[wave][nl][32 + hh * 16 + i];
  }

  // ---- flash state: one q-row per lane (q = qrow0 + nl, both halves agree) --
  float mrun = -__builtin_inff(), lrun = 0.f;
  v8f acc[4];                       // O^T tiles: acc[j][r] -> dv = j*16 + r + 8*hh
  #pragma unroll
  for (int j = 0; j < 4; ++j)
    #pragma unroll
    for (int e = 0; e < 8; ++e) acc[j][e] = 0.f;

  const int nkb_wg = 4 * qb + 4;    // WG-uniform causal block count
  // branchless prefetch assignment: half threads K, half V, rows via tid
  const float* pfbase = (tid & 1) ? vbase : kbase;
  const int    prow   = (tid >> 1) & 31;

  stage_kv(0, 0);
  __syncthreads();

  for (int kb = 0; kb < nkb_wg; ++kb) {
    const int key0 = kb * 32;
    const int cur  = kb & 1;

    if (kb + 1 < nkb_wg) {
      if (key0 + 64 < C_)   // gfx1250 global_prefetch_b8, two blocks ahead
        __builtin_prefetch(pfbase + (size_t)(key0 + 64 + prow) * ROWSTRIDE, 0, 1);
      stage_kv(cur ^ 1, key0 + 32);   // overlap staging with compute below
    }

    // wave-uniform causal activity: force a scalar branch (EXEC stays all-1s)
    if (__builtin_amdgcn_readfirstlane((key0 <= qrow0 + 15) ? 1 : 0)) {
      // ---- S^T tiles (16key x 16q): A = K tile, B = Q^T --------------------
      v16bf ak[4];                  // [tt*2+dh]
      #pragma unroll
      for (int tt = 0; tt < 2; ++tt)
        #pragma unroll
        for (int dh = 0; dh < 2; ++dh)
          #pragma unroll
          for (int i = 0; i < 16; ++i)
            ak[tt * 2 + dh][i] =
              lds_k[cur][tt * 16 + nl][dh * 32 + hh * 16 + i];

      v8f ST[2];
      #pragma unroll
      for (int tt = 0; tt < 2; ++tt) {
        v8f z;
        #pragma unroll
        for (int e = 0; e < 8; ++e) z[e] = 0.f;
        z = __builtin_amdgcn_wmma_f32_16x16x32_bf16(false, ak[tt * 2 + 0], false, bq0, (short)0, z, false, false);
        z = __builtin_amdgcn_wmma_f32_16x16x32_bf16(false, ak[tt * 2 + 1], false, bq1, (short)0, z, false, false);
        ST[tt] = z;
      }

      // ---- causal mask (only the diagonal-straddling block) ---------------
      float s0[8], s1[8];
      if (__builtin_amdgcn_readfirstlane((key0 + 31 <= qrow0) ? 1 : 0)) {
        #pragma unroll
        for (int r = 0; r < 8; ++r) { s0[r] = ST[0][r]; s1[r] = ST[1][r]; }
      } else {
        #pragma unroll
        for (int r = 0; r < 8; ++r) {
          const int krow = key0 + r + 8 * hh;          // this VGPR's key row
          s0[r] = (krow      <= qrow0 + nl) ? ST[0][r] : -__builtin_inff();
          s1[r] = (krow + 16 <= qrow0 + nl) ? ST[1][r] : -__builtin_inff();
        }
      }

      // ---- per-lane softmax over own 16 keys + one xor-16 combine ---------
      float bmax = fmaxf(s0[0], s1[0]);
      #pragma unroll
      for (int r = 1; r < 8; ++r) bmax = fmaxf(bmax, fmaxf(s0[r], s1[r]));
      bmax = fmaxf(bmax, SWZ(bmax, 0x401F));           // combine lane halves
      const float mnew  = fmaxf(mrun, bmax);
      const float scale = __expf(mrun - mnew);         // exp(-inf)=0 first block
      float e0[8], e1[8], esum = 0.f;
      #pragma unroll
      for (int r = 0; r < 8; ++r) {
        e0[r] = __expf(s0[r] - mnew);                  // masked: exp(-inf)=+0
        e1[r] = __expf(s1[r] - mnew);
        esum += e0[r] + e1[r];
      }
      esum += SWZ(esum, 0x401F);
      lrun = lrun * scale + esum;
      mrun = mnew;
      #pragma unroll
      for (int j = 0; j < 4; ++j) acc[j] *= scale;

      // ---- P^T B-layout via one xor-16 exchange (no LDS bounce) ----------
      // lane needs: hh=0 -> keys 0..7 own e0, keys 8..15 partner e0
      //             hh=1 -> keys 16..23 partner e1, keys 24..31 own e1
      v16bf bp;
      #pragma unroll
      for (int r = 0; r < 8; ++r) {
        const float sel  = hh ? e0[r] : e1[r];         // what partner needs
        const float recv = SWZ(sel, 0x401F);
        bp[r]     = (__bf16)(hh ? recv : e0[r]);
        bp[8 + r] = (__bf16)(hh ? e1[r] : recv);
      }

      // ---- O^T += V^T(16dv x 32key) x P^T(32key x 16q) per dv block -------
      v16bf av[4];
      #pragma unroll
      for (int j = 0; j < 4; ++j)
        #pragma unroll
        for (int i = 0; i < 16; ++i)
          av[j][i] = lds_vt[cur][j * 16 + nl][hh * 16 + i];
      #pragma unroll
      for (int j = 0; j < 4; ++j)
        acc[j] = __builtin_amdgcn_wmma_f32_16x16x32_bf16(false, av[j], false, bp, (short)0, acc[j], false, false);
    }

    __syncthreads();   // staging of kb+1 complete; reads of buffer `cur` done
  }

  // ---- normalize and store: lane owns q row (qrow0+nl), dv = j*16+8*hh+r ---
  const float inv = 1.0f / lrun;
  float* orow = outg + (((size_t)b * L_ + (size_t)n * C_ + qrow0 + nl) * H_ + h) * D_ + 8 * hh;
  #pragma unroll
  for (int j = 0; j < 4; ++j) {
    v4f o0, o1;
    #pragma unroll
    for (int r = 0; r < 4; ++r) { o0[r] = acc[j][r] * inv; o1[r] = acc[j][4 + r] * inv; }
    *(v4f*)&orow[j * 16]     = o0;
    *(v4f*)&orow[j * 16 + 4] = o1;
  }
}

extern "C" void kernel_launch(void* const* d_in, const int* in_sizes, int n_in,
                              void* d_out, int out_size, void* d_ws, size_t ws_size,
                              hipStream_t stream) {
  const float* q     = (const float*)d_in[0];
  const float* k     = (const float*)d_in[1];
  const float* v     = (const float*)d_in[2];
  const int*   start = (const int*)d_in[3];
  float*       out   = (float*)d_out;

  dim3 grid(B_ * NC_ * H_ * (C_ / (WAVES * 16)));   // 2048 workgroups
  dim3 block(WAVES * 32);                            // 256 threads = 8 waves
  hipLaunchKernelGGL(ChunkedAttention_23639499997831_kernel, grid, block, 0, stream,
                     q, k, v, start, out);
}
